// PhaseSyncAttention_69982197121389
// MI455X (gfx1250) — matrix operand order repaired and checked by reading, not compiled, not run on hardware
//
#include <hip/hip_runtime.h>
#include <hip/hip_bf16.h>

typedef __attribute__((ext_vector_type(16))) _Float16 v16h;
typedef __attribute__((ext_vector_type(8)))  _Float16 v8h;
typedef __attribute__((ext_vector_type(8)))  float    v8f;

#define B_  4
#define L_  1024
#define H_  8
#define E_  64
#define C_  512          // H*E
#define SCALE_ 0.125f    // 1/sqrt(64)

// ---------------- WMMA helpers (CDNA5 v_wmma_f32_16x16x32_f16) ----------------

__device__ __forceinline__ v8f wmma32(v16h a, v16h b, v8f c) {
    return __builtin_amdgcn_wmma_f32_16x16x32_f16(
        /*neg_a=*/false, a, /*neg_b=*/false, b,
        /*c_mod=*/(short)0, c, /*reuse_a=*/false, /*reuse_b=*/false);
}

__device__ __forceinline__ v16h make_frag(const _Float16* p0, const _Float16* p1) {
    v8h x = *(const v8h*)p0;
    v8h y = *(const v8h*)p1;
    v16h r;
#pragma unroll
    for (int i = 0; i < 8; ++i) { r[i] = x[i]; r[i + 8] = y[i]; }
    return r;
}

// A fragment: rows M=0..15 (row-major in LDS, stride ldm halves), K in [kbase,kbase+32)
// ISA layout: lanes 0-15 M=lane, halves j<8 -> K=kbase+hi*8+j ; j>=8 -> K=kbase+16+hi*8+(j-8)
__device__ __forceinline__ v16h load_a_frag(const _Float16* base, int ldm, int kbase,
                                            int lo, int hi) {
    const _Float16* p = base + lo * ldm + kbase + hi * 8;
    return make_frag(p, p + 16);
}

// B fragment: element (k,n): n = lo (column), j<8 -> k = hi*8 + j ; j>=8 -> k = 16 + hi*8 + (j-8).
// rowptr must already point at (column lo)'s row with kbase applied.
__device__ __forceinline__ v16h load_b_frag(const _Float16* rowptr, int hi) {
    const _Float16* p = rowptr + hi * 8;
    return make_frag(p, p + 16);
}

// ---------------- CDNA5 async global->LDS copy (ASYNCcnt tracked) ----------------

__device__ __forceinline__ void async_copy16(const _Float16* gsrc, _Float16* ldst) {
    unsigned lds_off = (unsigned)(size_t)ldst;              // low 32 bits = LDS byte addr
    unsigned long long ga = (unsigned long long)(size_t)gsrc;
    asm volatile("global_load_async_to_lds_b128 %0, %1, off"
                 :: "v"(lds_off), "v"(ga) : "memory");
}

__device__ __forceinline__ void async_wait0() {
    asm volatile("s_wait_asynccnt 0x0" ::: "memory");
}

// ---------------- K0: Hilbert time-domain filter ----------------
// g_im[d] = (2/L) * sum_{k=1}^{L/2-1} sin(2*pi*k*d/L); exactly 0 for even d.
__global__ __launch_bounds__(256) void k_hilbert(float* hilb) {
    int d = blockIdx.x * 256 + threadIdx.x;
    if (d >= L_) return;
    float s = 0.f;
    if (d & 1) {
        for (int k = 1; k <= L_ / 2 - 1; ++k) {
            int kd = (k * d) & (L_ - 1);             // exact mod-L reduction
            s += sinf(6.283185307179586f * (float)kd * (1.0f / (float)L_));
        }
        s *= (2.0f / (float)L_);
    }
    hilb[d] = s;
}

// ---------------- K1: activations ----------------
__global__ __launch_bounds__(256) void k_act(const float* __restrict__ hurst,
                                             const float* __restrict__ tau,
                                             float* __restrict__ hexp,
                                             float* __restrict__ taut) {
    int i = blockIdx.x * 256 + threadIdx.x;          // B*L*H = 32768
    float hs = 1.0f / (1.0f + expf(-hurst[i]));
    float ht = hs * 0.99f + 0.01f;
    hexp[i] = expf(2.0f * ht);                       // GAMMA = 2
    float ts = 1.0f / (1.0f + expf(-tau[i]));
    taut[i] = ts * 0.9f + 0.1f;
}

// ---------------- K2: inclusive cumsum of hexp along L per (b,h) ----------------
__global__ __launch_bounds__(1024) void k_scan(const float* __restrict__ hexp,
                                               float* __restrict__ psi) {
    __shared__ float buf[L_];
    int bh = blockIdx.x;                             // B*H = 32
    int b = bh >> 3, h = bh & 7;
    int t = threadIdx.x;
    int idx = (b * L_ + t) * H_ + h;
    buf[t] = hexp[idx];
    __syncthreads();
    for (int off = 1; off < L_; off <<= 1) {
        float add = (t >= off) ? buf[t - off] : 0.f;
        __syncthreads();
        buf[t] += add;
        __syncthreads();
    }
    psi[idx] = buf[t];
}

// ---------------- K3: Hilbert circular convolution per (b, channel) ----------------
// Quirky reshape: signal position p = h*128 + l/8 ; channel c = (l%8)*64 + e.
__global__ __launch_bounds__(256) void k_hilx(const float* __restrict__ q,
                                              const float* __restrict__ hilb,
                                              float* __restrict__ sinb,
                                              float* __restrict__ cosb) {
    __shared__ float xl[L_];
    __shared__ float hl[L_];
    int bc = blockIdx.x;                             // B*C = 2048
    int b = bc >> 9;
    int c = bc & (C_ - 1);
    int lmod = c >> 6;                               // l % 8
    int e = c & 63;
    int t = threadIdx.x;
#pragma unroll
    for (int i = 0; i < 4; ++i) {
        int p = i * 256 + t;
        int h = p >> 7;
        int l = ((p & 127) << 3) | lmod;
        xl[p] = q[((b * L_ + l) * H_ + h) * E_ + e];
        hl[p] = hilb[p];
    }
    __syncthreads();
#pragma unroll 1
    for (int i = 0; i < 4; ++i) {
        int p = i * 256 + t;
        float hx = 0.f;
        for (int d = 1; d < L_; d += 2)
            hx += xl[(p - d) & (L_ - 1)] * hl[d];
        float xv = xl[p];
        float r = sqrtf(xv * xv + hx * hx);
        float sp, cp;
        if (r > 0.f) { sp = hx / r; cp = xv / r; }
        else         { sp = 0.f;   cp = 1.f;   }    // atan2(0,0)=0
        size_t o = (size_t)bc * L_ + p;
        sinb[o] = sp;
        cosb[o] = cp;
    }
}

// ---------------- K4: theta = atan2(sum_c sin, sum_c cos) ----------------
__global__ __launch_bounds__(128) void k_theta(const float* __restrict__ sinb,
                                               const float* __restrict__ cosb,
                                               float* __restrict__ theta) {
    __shared__ float rs[128], rc[128];
    int bl = blockIdx.x;                             // B*L = 4096
    int b = bl >> 10, l = bl & (L_ - 1);
    int t = threadIdx.x;
    float ss = 0.f, cc = 0.f;
#pragma unroll
    for (int i = 0; i < 4; ++i) {
        int c = i * 128 + t;
        size_t o = ((size_t)(b * C_ + c)) * L_ + l;
        ss += sinb[o];
        cc += cosb[o];
    }
    rs[t] = ss; rc[t] = cc;
    __syncthreads();
    for (int off = 64; off > 0; off >>= 1) {
        if (t < off) { rs[t] += rs[t + off]; rc[t] += rc[t + off]; }
        __syncthreads();
    }
    if (t == 0) theta[bl] = atan2f(rs[0], rc[0]);
}

// ---------------- K5: repack to f16: Qh/Kh [bh][l][e], Vt [bh][e][l] ----------------
__global__ __launch_bounds__(256) void k_repack(const float* __restrict__ q,
                                                const float* __restrict__ k,
                                                const float* __restrict__ v,
                                                _Float16* __restrict__ Qh,
                                                _Float16* __restrict__ Kh,
                                                _Float16* __restrict__ Vt) {
    int i = blockIdx.x * 256 + threadIdx.x;          // B*L*H*E = 2,097,152
    int e = i & 63, h = (i >> 6) & 7, l = (i >> 9) & (L_ - 1), b = i >> 19;
    int bh = b * H_ + h;
    Qh[((size_t)bh * L_ + l) * E_ + e] = (_Float16)q[i];
    Kh[((size_t)bh * L_ + l) * E_ + e] = (_Float16)k[i];
    Vt[((size_t)bh * E_ + e) * L_ + l] = (_Float16)v[i];
}

// ---------------- K6: fused attention (WMMA QK^T + stats + WMMA PV) ----------------
__global__ __launch_bounds__(256) void k_attn(const _Float16* __restrict__ Qh,
                                              const _Float16* __restrict__ Kh,
                                              const _Float16* __restrict__ Vt,
                                              const float* __restrict__ psi,
                                              const float* __restrict__ taut,
                                              const float* __restrict__ theta,
                                              float* __restrict__ out) {
    __shared__ __align__(16) _Float16 sw[16 * L_];   // scores -> weights (f16), 32KB
    __shared__ __align__(16) _Float16 q2[16 * 72];   // Q tile, padded
    __shared__ __align__(16) _Float16 kv[128 * 72];  // K chunk / psi+theta / 2x V chunk
    __shared__ float rowred[3][16][16];
    __shared__ float rowstat[16][3];
    __shared__ float pvred[4][16][16];               // PV cross-wave combine

    const int wg = blockIdx.x;                       // 2048 = B*H*64
    const int rb = wg & 63;
    const int bh = wg >> 6;
    const int b = bh >> 3, h = bh & 7;
    const int l0 = rb * 16;
    const int t = threadIdx.x;
    const int lane = t & 31, wave = t >> 5;
    const int lo = lane & 15, hi = lane >> 4;

    // ---- load Q tile (16 rows x 64 halves) ----
    if (t < 128) {
        int row = t >> 3, c8 = t & 7;
        *(uint4*)(q2 + row * 72 + c8 * 8) =
            *(const uint4*)(Qh + ((size_t)bh * L_ + l0 + row) * E_ + c8 * 8);
    }
    __syncthreads();

    const v16h aq0 = load_a_frag(q2, 72, 0,  lo, hi);
    const v16h aq1 = load_a_frag(q2, 72, 32, lo, hi);

    // ---- QK^T: 8 chunks of 128 columns; each wave owns one 16x16 tile per chunk ----
    for (int sc = 0; sc < 8; ++sc) {
#pragma unroll
        for (int i = 0; i < 4; ++i) {                // async global -> LDS staging
            int idx = i * 256 + t;
            int row = idx >> 3, c8 = idx & 7;
            async_copy16(Kh + ((size_t)bh * L_ + sc * 128 + row) * E_ + c8 * 8,
                         kv + row * 72 + c8 * 8);
        }
        async_wait0();
        __syncthreads();
        const _Float16* krow = kv + (wave * 16 + lo) * 72;
        v8f c = {0.f, 0.f, 0.f, 0.f, 0.f, 0.f, 0.f, 0.f};
        c = wmma32(aq0, load_b_frag(krow + 0,  hi), c);
        c = wmma32(aq1, load_b_frag(krow + 32, hi), c);
        int col = sc * 128 + wave * 16 + lo;
#pragma unroll
        for (int r8 = 0; r8 < 8; ++r8)
            sw[(hi * 8 + r8) * L_ + col] = (_Float16)(c[r8] * SCALE_);
        __syncthreads();
    }

    // ---- stage per-column psi/theta into LDS (reuse kv) ----
    float* psic = (float*)kv;
    float* thec = psic + L_;
#pragma unroll
    for (int i = 0; i < 4; ++i) {
        int s = i * 256 + t;
        psic[s] = psi[((size_t)b * L_ + s) * H_ + h];
        thec[s] = theta[b * L_ + s];
    }
    __syncthreads();

    // ---- row stats + weight combine: thread (r,seg) owns 64 columns of row r ----
    {
        const int r = t >> 4, seg = t & 15;
        const int lrow = l0 + r;
        const float psi_r = psi[((size_t)b * L_ + lrow) * H_ + h];
        const float th_r  = theta[b * L_ + lrow];
        const float tau_r = taut[((size_t)b * L_ + lrow) * H_ + h];
        const float inv2t2 = 1.0f / (2.0f * tau_r * tau_r);
        _Float16* swr = sw + r * L_ + seg * 64;

        // pass 1: row max
        float m = -3.0e38f;
        for (int j = 0; j < 64; ++j) m = fmaxf(m, (float)swr[j]);
        rowred[0][r][seg] = m;
        __syncthreads();
        if (seg == 0) {
            float mm = -3.0e38f;
            for (int i = 0; i < 16; ++i) mm = fmaxf(mm, rowred[0][r][i]);
            rowstat[r][0] = mm;
        }
        __syncthreads();
        const float mrow = rowstat[r][0];

        // pass 2: sum exp, sum prior, sum prior*gate
        float se = 0.f, sp1 = 0.f, spg = 0.f;
        for (int j = 0; j < 64; ++j) {
            int s = seg * 64 + j;
            float a = (float)swr[j];
            se += expf(a - mrow);
            float d = psi_r - psic[s];
            float p1 = expf(-0.5f * d * d);                    // SIGMA_G = 1
            float dphi = sinf(0.5f * (th_r - thec[s]));
            float g = expf(-dphi * dphi * inv2t2);
            sp1 += p1;
            spg += p1 * g;
        }
        rowred[0][r][seg] = se;
        rowred[1][r][seg] = sp1;
        rowred[2][r][seg] = spg;
        __syncthreads();
        if (seg == 0) {
            float s0 = 0.f, s1 = 0.f, s2 = 0.f;
            for (int i = 0; i < 16; ++i) {
                s0 += rowred[0][r][i]; s1 += rowred[1][r][i]; s2 += rowred[2][r][i];
            }
            float n1 = s1 + 1e-6f;                // first normalization
            float n2 = s2 / n1 + 1e-6f;           // second normalization (post-gate)
            rowstat[r][1] = 0.9f / s0;            // (1-ALPHA)/sumexp
            rowstat[r][2] = 0.1f / (n1 * n2);     // ALPHA/(n1*n2)
        }
        __syncthreads();
        const float cA = rowstat[r][1], cP = rowstat[r][2];

        // pass 3: combined weight, in-place f16
        for (int j = 0; j < 64; ++j) {
            int s = seg * 64 + j;
            float a = (float)swr[j];
            float d = psi_r - psic[s];
            float p1 = expf(-0.5f * d * d);
            float dphi = sinf(0.5f * (th_r - thec[s]));
            float g = expf(-dphi * dphi * inv2t2);
            float w = cA * expf(a - mrow) + cP * p1 * g;
            swr[j] = (_Float16)w;
        }
        __syncthreads();
    }

    // ---- PV: out[16 x 64] = W[16 x 1024] * V[1024 x 64] ----
    // 8 waves: wave = shalf*4 + etile. kv double-buffered: two 64x72 V chunks.
    const int et = wave & 3, sh = wave >> 2;
    v8f acc = {0.f, 0.f, 0.f, 0.f, 0.f, 0.f, 0.f, 0.f};
    for (int vc = 0; vc < 8; ++vc) {
#pragma unroll
        for (int i = 0; i < 4; ++i) {                // stage both s-halves async
            int idx = i * 256 + t;
            int half = idx >> 9, j = idx & 511;
            int row = j >> 3, c8 = j & 7;            // row = e, c8*8 = s offset in chunk
            async_copy16(Vt + ((size_t)bh * E_ + row) * L_ + half * 512 + vc * 64 + c8 * 8,
                         kv + half * (64 * 72) + row * 72 + c8 * 8);
        }
        async_wait0();
        __syncthreads();
#pragma unroll
        for (int ks = 0; ks < 2; ++ks) {
            v16h aw = load_a_frag(sw, L_, sh * 512 + vc * 64 + ks * 32, lo, hi);
            v16h bv = load_b_frag(kv + sh * (64 * 72) + (et * 16 + lo) * 72 + ks * 32, hi);
            acc = wmma32(aw, bv, acc);
        }
        __syncthreads();
    }
    // combine s-halves: waves 4-7 publish, waves 0-3 add + store
    if (sh == 1) {
#pragma unroll
        for (int r8 = 0; r8 < 8; ++r8)
            pvred[et][hi * 8 + r8][lo] = acc[r8];
    }
    __syncthreads();
    if (sh == 0) {
#pragma unroll
        for (int r8 = 0; r8 < 8; ++r8) {
            int l = l0 + hi * 8 + r8;
            out[((size_t)(b * L_ + l) * H_ + h) * E_ + et * 16 + lo] =
                acc[r8] + pvred[et][hi * 8 + r8][lo];
        }
    }
}

// ---------------- host launcher ----------------
extern "C" void kernel_launch(void* const* d_in, const int* in_sizes, int n_in,
                              void* d_out, int out_size, void* d_ws, size_t ws_size,
                              hipStream_t stream) {
    (void)in_sizes; (void)n_in; (void)out_size; (void)ws_size;

    const float* q     = (const float*)d_in[0];
    const float* k     = (const float*)d_in[1];
    const float* v     = (const float*)d_in[2];
    // d_in[3] = sigma : unused by the reference forward
    const float* hurst = (const float*)d_in[4];
    const float* tau   = (const float*)d_in[5];
    float* out = (float*)d_out;

    char* ws = (char*)d_ws;
    size_t off = 0;
    auto alloc = [&](size_t bytes) -> char* {
        char* p = ws + off;
        off += (bytes + 255) & ~(size_t)255;
        return p;
    };

    float*    hilb = (float*)alloc((size_t)L_ * 4);
    float*    hexp = (float*)alloc((size_t)B_ * L_ * H_ * 4);
    float*    taut = (float*)alloc((size_t)B_ * L_ * H_ * 4);
    float*    psi  = (float*)alloc((size_t)B_ * L_ * H_ * 4);
    float*    thet = (float*)alloc((size_t)B_ * L_ * 4);
    float*    sinb = (float*)alloc((size_t)B_ * C_ * L_ * 4);
    float*    cosb = (float*)alloc((size_t)B_ * C_ * L_ * 4);
    _Float16* Qh   = (_Float16*)alloc((size_t)B_ * H_ * L_ * E_ * 2);
    _Float16* Kh   = (_Float16*)alloc((size_t)B_ * H_ * L_ * E_ * 2);
    _Float16* Vt   = (_Float16*)alloc((size_t)B_ * H_ * L_ * E_ * 2);

    k_hilbert<<<(L_ + 255) / 256, 256, 0, stream>>>(hilb);
    k_act<<<(B_ * L_ * H_) / 256, 256, 0, stream>>>(hurst, tau, hexp, taut);
    k_scan<<<B_ * H_, L_, 0, stream>>>(hexp, psi);
    k_hilx<<<B_ * C_, 256, 0, stream>>>(q, hilb, sinb, cosb);
    k_theta<<<B_ * L_, 128, 0, stream>>>(sinb, cosb, thet);
    k_repack<<<(B_ * L_ * H_ * E_) / 256, 256, 0, stream>>>(q, k, v, Qh, Kh, Vt);
    k_attn<<<B_ * H_ * (L_ / 16), 256, 0, stream>>>(Qh, Kh, Vt, psi, taut, thet, out);
}
